// EmbedMatcher_84095459656274
// MI455X (gfx1250) — compile-verified, matrix operand order checked by reading, not run on hardware
//
#include <hip/hip_runtime.h>
#include <hip/hip_bf16.h>
#include <math.h>

typedef __attribute__((ext_vector_type(16))) _Float16 v16h;
typedef __attribute__((ext_vector_type(8)))  float    v8f;

#define EMBD 128
#define DM   256
#define HID  512
#define GW   2048
#define BQ   1024
#define SS   5
#define KN   256
#define PADID 200000

// async copy global -> LDS, 16B per lane, tracked by ASYNCcnt
#define ASYNC_B128(ldsoff, gsrc)                                        \
  asm volatile("global_load_async_to_lds_b128 %0, %1, off"              \
               :: "v"(ldsoff), "v"(gsrc) : "memory")
#define WAIT_ASYNC() asm volatile("s_wait_asynccnt 0x0" ::: "memory")

// ---------------- helpers ----------------

__device__ __forceinline__ float sigm(float x) { return 1.0f / (1.0f + expf(-x)); }

__device__ __forceinline__ float block_sum256(float v, float* red) {
  int t = threadIdx.x;
  red[t] = v; __syncthreads();
#pragma unroll
  for (int s = 128; s > 0; s >>= 1) {
    if (t < s) red[t] += red[t + s];
    __syncthreads();
  }
  float r = red[0]; __syncthreads();
  return r;
}

// A fragment: 16x32 f16, row-major source, documented CDNA5 layout:
// lane<16: row=lane, K = {0..7,16..23}; lane>=16: row=lane-16, K={8..15,24..31}
__device__ __forceinline__ v16h load_a_frag(const _Float16* A, int m0, int kb,
                                            int M, int K, int lane) {
  v16h a;
  int row = m0 + (lane & 15);
  int kadd = (lane < 16) ? 0 : 8;
  bool ok = (row < M);
  const _Float16* ap = A + (size_t)row * K + kb + kadd;
#pragma unroll
  for (int e = 0; e < 16; ++e) {
    int k = (e < 8) ? e : (e + 8);
    a[e] = ok ? ap[k] : (_Float16)0.0f;
  }
  return a;
}

__device__ __forceinline__ v16h load_b_frag(const _Float16* Wt, int n0, int kb,
                                            int N, int K, int lane) {
  v16h b;
  int col = n0 + (lane & 15);
  int kadd = (lane < 16) ? 0 : 8;
  bool ok = (col < N);
  const _Float16* bp = Wt + (size_t)col * K + kb + kadd;
#pragma unroll
  for (int e = 0; e < 16; ++e) {
    int k = (e < 8) ? e : (e + 8);
    b[e] = ok ? bp[k] : (_Float16)0.0f;
  }
  return b;
}

// ---------------- tiny prep kernels ----------------

__global__ void k_cvt_f16(const float* __restrict__ src, _Float16* __restrict__ dst, int n) {
  int i = blockIdx.x * blockDim.x + threadIdx.x;
  if (i < n) dst[i] = (_Float16)src[i];
}

__global__ void k_add_bias(const float* a, const float* b, float* o, int n) {
  int i = blockIdx.x * blockDim.x + threadIdx.x;
  if (i < n) o[i] = a[i] + b[i];
}

__global__ void k_zero_u32(unsigned int* p, int n) {
  int i = blockIdx.x * blockDim.x + threadIdx.x;
  if (i < n) p[i] = 0u;
}

// ---------------- neighbor encoder ----------------
// One block (8 wave32) per row: stream gathers for similarity, LDS top-10,
// then WMMA-project only the 10 selected rows (96% of reference GEMM skipped).
__global__ __launch_bounds__(256)
void k_neighbor(const int* __restrict__ conn, const int* __restrict__ ids, int id_col,
                const float* __restrict__ emb, const _Float16* __restrict__ gcnh,
                const float* __restrict__ gcn_wb, const float* __restrict__ gcn_bb,
                const float* __restrict__ gate_ww, const float* __restrict__ gate_wb,
                const float* __restrict__ gate_bb,
                float* __restrict__ outf, _Float16* __restrict__ outh,
                int col_off) {
  int b = blockIdx.x;
  int t = threadIdx.x;
  int lane = t & 31;
  int wave = t >> 5;

  __shared__ float selfv[EMBD];
  __shared__ float qrelv[EMBD];
  __shared__ float simb[KN];
  __shared__ unsigned char padb[KN];
  __shared__ float redf[256];
  __shared__ int   redi[256];
  __shared__ int   selid[10];
  __shared__ float aggp[256];
  __shared__ float aggv[EMBD];
  __shared__ float sc[4];
  __shared__ _Float16 nb[16 * DM];

  const int selfid = ids[b * 3 + id_col];
  const int qrelid = ids[b * 3 + 2];
  if (t < EMBD) selfv[t] = emb[(size_t)selfid * EMBD + t];
  else          qrelv[t - EMBD] = emb[(size_t)qrelid * EMBD + (t - EMBD)];
  __syncthreads();

  float p = (t < EMBD) ? selfv[t] * selfv[t] : 0.0f;
  float ns = block_sum256(p, redf);
  p = (t < EMBD) ? qrelv[t] * qrelv[t] : 0.0f;
  float nq = block_sum256(p, redf);
  if (t == 0) {
    sc[0] = 1.0f / fmaxf(sqrtf(ns), 1e-8f);
    sc[1] = 1.0f / fmaxf(sqrtf(nq), 1e-8f);
  }
  __syncthreads();

  {
    float inv_s = sc[0], inv_q = sc[1];
    int rel = conn[((size_t)b * KN + t) * 2 + 0];
    int ent = conn[((size_t)b * KN + t) * 2 + 1];
    const float* ep = emb + (size_t)ent * EMBD;
    const float* rp = emb + (size_t)rel * EMBD;
    float de = 0.f, ne = 0.f, dr = 0.f, nr = 0.f;
    for (int j = 0; j < EMBD; ++j) {
      float ev = ep[j]; de += selfv[j] * ev; ne += ev * ev;
      float rv = rp[j]; dr += qrelv[j] * rv; nr += rv * rv;
    }
    float se = de * inv_s / fmaxf(sqrtf(ne), 1e-8f);
    float sr = dr * inv_q / fmaxf(sqrtf(nr), 1e-8f);
    float s = 0.7f * se + 0.3f * sr;
    bool isp = (rel == PADID);
    simb[t] = isp ? -INFINITY : s;
    padb[t] = isp ? 1 : 0;
  }
  __syncthreads();

  int cnt = 0;
  for (int it = 0; it < 10; ++it) {
    redf[t] = simb[t]; redi[t] = t; __syncthreads();
    for (int s2 = 128; s2 > 0; s2 >>= 1) {
      if (t < s2) {
        if (redf[t + s2] > redf[t]) { redf[t] = redf[t + s2]; redi[t] = redi[t + s2]; }
      }
      __syncthreads();
    }
    if (t == 0) {
      int idx = redi[0];
      selid[it] = idx;
      simb[idx] = -INFINITY;
      if (!padb[idx]) cnt++;
    }
    __syncthreads();
  }
  if (t == 0) sc[2] = (float)cnt;
  __syncthreads();

  for (int idx = t; idx < 16 * DM; idx += 256) {
    int row = idx >> 8, col = idx & 255;
    _Float16 v = (_Float16)0.0f;
    if (row < 10) {
      int kk = selid[row];
      int id = conn[((size_t)b * KN + kk) * 2 + ((col < EMBD) ? 0 : 1)];
      v = (_Float16)emb[(size_t)id * EMBD + (col & 127)];
    }
    nb[idx] = v;
  }
  __syncthreads();

  v8f acc = {0.f, 0.f, 0.f, 0.f, 0.f, 0.f, 0.f, 0.f};
  int n0 = wave << 4;
  int rowA = lane & 15;
  int kadd = (lane < 16) ? 0 : 8;
  int colB = n0 + (lane & 15);
  for (int kb = 0; kb < DM; kb += 32) {
    v16h a, bq;
#pragma unroll
    for (int e = 0; e < 16; ++e) {
      int k = kb + kadd + ((e < 8) ? e : (e + 8));
      a[e]  = nb[rowA * DM + k];
      bq[e] = gcnh[(size_t)colB * DM + k];
    }
    acc = __builtin_amdgcn_wmma_f32_16x16x32_f16(false, a, false, bq,
                                                 (short)0, acc, false, false);
  }

  float bn = gcn_wb[colB] + gcn_bb[colB];
  float psum = 0.0f;
  int mb = (lane < 16) ? 0 : 8;
#pragma unroll
  for (int r = 0; r < 8; ++r) {
    int m = mb + r;
    if (m < 10) {
      float v = acc[r] + bn;
      v = (v > 0.0f) ? v : 0.01f * v;
      psum += v;
    }
  }
  aggp[t] = psum;
  __syncthreads();
  if (lane < 16) {
    float a2 = aggp[wave * 32 + lane] + aggp[wave * 32 + lane + 16];
    aggv[n0 + lane] = a2 / (sc[2] + 1e-9f);
  }
  __syncthreads();

  p = (t < EMBD) ? aggv[t] * gate_ww[t] : 0.0f;
  float gs = block_sum256(p, redf);
  if (t == 0) sc[3] = sigm(gs + gate_wb[0] + gate_bb[0]);
  __syncthreads();
  if (t < EMBD) {
    float g = sc[3];
    float o = tanhf(g * aggv[t] + (1.0f - g) * selfv[t]);
    size_t oi = (size_t)b * DM + col_off + t;
    outf[oi] = o;
    outh[oi] = (_Float16)o;
  }
}

// ---------------- small direct WMMA GEMM (ragged M, support rows) ----------------
__global__ __launch_bounds__(256)
void k_gemm_wmma(const _Float16* __restrict__ A, const _Float16* __restrict__ Wt,
                 const float* __restrict__ bias, const float* __restrict__ Cadd,
                 float* __restrict__ Cout, _Float16* __restrict__ CoutH,
                 int M, int N, int K, int relu) {
  int lane = threadIdx.x & 31;
  int wave = threadIdx.x >> 5;
  int Nt = (N + 15) >> 4;
  int Mt = (M + 15) >> 4;
  int tid = blockIdx.x * 8 + wave;
  if (tid >= Mt * Nt) return;
  int mt = tid / Nt, nt = tid - mt * Nt;
  int m0 = mt << 4, n0 = nt << 4;

  v8f acc = {0.f, 0.f, 0.f, 0.f, 0.f, 0.f, 0.f, 0.f};
  for (int kb = 0; kb < K; kb += 32) {
    v16h a = load_a_frag(A, m0, kb, M, K, lane);
    v16h b = load_b_frag(Wt, n0, kb, N, K, lane);
    acc = __builtin_amdgcn_wmma_f32_16x16x32_f16(false, a, false, b,
                                                 (short)0, acc, false, false);
  }

  int n = n0 + (lane & 15);
  if (n >= N) return;
  float bn = bias ? bias[n] : 0.0f;
  int mbase = m0 + ((lane < 16) ? 0 : 8);
#pragma unroll
  for (int r = 0; r < 8; ++r) {
    int m = mbase + r;
    if (m < M) {
      float v = acc[r] + bn;
      if (Cadd) v += Cadd[(size_t)m * N + n];
      if (relu) v = (v > 0.0f) ? v : 0.0f;
      if (Cout)  Cout[(size_t)m * N + n] = v;
      if (CoutH) CoutH[(size_t)m * N + n] = (_Float16)v;
    }
  }
}

// ---------------- large LDS-staged WMMA GEMM ----------------
// 128x64 block tile, BK=32, double-buffered GLOBAL_LOAD_ASYNC_TO_LDS_B128
// staging (ASYNCcnt pipeline). Requires M%128==0, N%64==0, K%32==0.
// 8 waves: wave w computes rows [16w,16w+16) x all 64 N columns (4 WMMAs,
// A fragment reused 4x from registers, B tile shared across waves via LDS).
__global__ __launch_bounds__(256)
void k_gemm_wmma_lds(const _Float16* __restrict__ A, const _Float16* __restrict__ Wt,
                     const float* __restrict__ bias, const float* __restrict__ Cadd,
                     float* __restrict__ Cout, _Float16* __restrict__ CoutH,
                     int M, int N, int K, int relu) {
  __shared__ _Float16 As[2][128 * 32];
  __shared__ _Float16 Bs[2][64 * 32];
  int t = threadIdx.x;
  int lane = t & 31;
  int wave = t >> 5;
  int Nb = N >> 6;
  int mb = blockIdx.x / Nb;
  int nb = blockIdx.x - mb * Nb;
  int m0b = mb << 7;
  int n0b = nb << 6;

  // staging coords: 16B chunk per issue; A: 512 chunks (2/thread), B: 256 (1/thread)
  int arow0 = t >> 2;
  int arow1 = (t + 256) >> 2;
  int brow  = t >> 2;
  int segH  = (t & 3) << 3;   // half-element offset within row (0,8,16,24)
  uint32_t segB = (uint32_t)((t & 3) << 4); // byte offset (0,16,32,48)

  uint32_t asb0 = (uint32_t)(uintptr_t)&As[0][0];
  uint32_t asb1 = (uint32_t)(uintptr_t)&As[1][0];
  uint32_t bsb0 = (uint32_t)(uintptr_t)&Bs[0][0];
  uint32_t bsb1 = (uint32_t)(uintptr_t)&Bs[1][0];

  const _Float16* a0p = A + (size_t)(m0b + arow0) * K + segH;
  const _Float16* a1p = A + (size_t)(m0b + arow1) * K + segH;
  const _Float16* bp  = Wt + (size_t)(n0b + brow) * K + segH;
  uint32_t da0 = (uint32_t)(arow0 * 64) + segB;
  uint32_t da1 = (uint32_t)(arow1 * 64) + segB;
  uint32_t db  = (uint32_t)(brow * 64) + segB;

  int nk = K >> 5;
  // prologue: stage k-step 0 into buffer 0
  ASYNC_B128(asb0 + da0, a0p);
  ASYNC_B128(asb0 + da1, a1p);
  ASYNC_B128(bsb0 + db,  bp);

  v8f acc0 = {0.f,0.f,0.f,0.f,0.f,0.f,0.f,0.f};
  v8f acc1 = acc0, acc2 = acc0, acc3 = acc0;

  int rowA = (wave << 4) + (lane & 15);
  int kadd = (lane < 16) ? 0 : 8;
  int buf = 0;
  for (int s = 0; s < nk; ++s) {
    WAIT_ASYNC();
    __syncthreads();
    if (s + 1 < nk) {
      int kb = (s + 1) << 5;
      uint32_t ab = buf ? asb0 : asb1;
      uint32_t bb = buf ? bsb0 : bsb1;
      ASYNC_B128(ab + da0, a0p + kb);
      ASYNC_B128(ab + da1, a1p + kb);
      ASYNC_B128(bb + db,  bp + kb);
    }
    const _Float16* as = &As[buf][0];
    const _Float16* bs = &Bs[buf][0];
    v16h a;
#pragma unroll
    for (int e = 0; e < 16; ++e) {
      int k = kadd + ((e < 8) ? e : (e + 8));
      a[e] = as[rowA * 32 + k];
    }
    v16h bf0, bf1, bf2, bf3;
#pragma unroll
    for (int e = 0; e < 16; ++e) {
      int k = kadd + ((e < 8) ? e : (e + 8));
      int c = lane & 15;
      bf0[e] = bs[(c) * 32 + k];
      bf1[e] = bs[(16 + c) * 32 + k];
      bf2[e] = bs[(32 + c) * 32 + k];
      bf3[e] = bs[(48 + c) * 32 + k];
    }
    acc0 = __builtin_amdgcn_wmma_f32_16x16x32_f16(false, a, false, bf0, (short)0, acc0, false, false);
    acc1 = __builtin_amdgcn_wmma_f32_16x16x32_f16(false, a, false, bf1, (short)0, acc1, false, false);
    acc2 = __builtin_amdgcn_wmma_f32_16x16x32_f16(false, a, false, bf2, (short)0, acc2, false, false);
    acc3 = __builtin_amdgcn_wmma_f32_16x16x32_f16(false, a, false, bf3, (short)0, acc3, false, false);
    buf ^= 1;
  }

  int n_lane = lane & 15;
  int mbase = m0b + (wave << 4) + ((lane < 16) ? 0 : 8);
#pragma unroll
  for (int ns2 = 0; ns2 < 4; ++ns2) {
    v8f accv = (ns2 == 0) ? acc0 : (ns2 == 1) ? acc1 : (ns2 == 2) ? acc2 : acc3;
    int n = n0b + (ns2 << 4) + n_lane;
    float bn = bias ? bias[n] : 0.0f;
#pragma unroll
    for (int r = 0; r < 8; ++r) {
      int m = mbase + r;
      float v = accv[r] + bn;
      if (Cadd) v += Cadd[(size_t)m * N + n];
      if (relu) v = (v > 0.0f) ? v : 0.0f;
      if (Cout)  Cout[(size_t)m * N + n] = v;
      if (CoutH) CoutH[(size_t)m * N + n] = (_Float16)v;
    }
  }
}

// ---------------- layernorm / mean / cell / final ----------------
__global__ __launch_bounds__(256)
void k_layernorm(const float* __restrict__ Y, const float* __restrict__ g,
                 const float* __restrict__ bb, float* __restrict__ Out,
                 _Float16* __restrict__ OutH) {
  __shared__ float red[256];
  int b = blockIdx.x, t = threadIdx.x;
  float y = Y[(size_t)b * DM + t];
  float mu = block_sum256(y, red) * (1.0f / DM);
  float dv = y - mu;
  float var = block_sum256(dv * dv, red) * (1.0f / DM);
  float o = g[t] * dv * rsqrtf(var + 1e-5f) + bb[t];
  Out[(size_t)b * DM + t] = o;
  if (OutH) OutH[(size_t)b * DM + t] = (_Float16)o;
}

__global__ void k_support_mean(const float* __restrict__ senc,
                               float* __restrict__ sg, _Float16* __restrict__ sgh) {
  int t = threadIdx.x;
  float s = 0.0f;
  for (int r = 0; r < SS; ++r) s += senc[r * DM + t];
  s *= (1.0f / SS);
  sg[t] = s;
  sgh[t] = (_Float16)s;
}

__global__ __launch_bounds__(512)
void k_lstm_cell(const float* __restrict__ gates, const float* __restrict__ qenc,
                 const float* __restrict__ sg, float* __restrict__ c,
                 float* __restrict__ h, _Float16* __restrict__ hr) {
  int b = blockIdx.x, j = threadIdx.x;
  const float* gb = gates + (size_t)b * GW;
  float gi = gb[j], gf = gb[HID + j], gg = gb[2 * HID + j], go = gb[3 * HID + j];
  float cv = c[(size_t)b * HID + j];
  cv = sigm(gf) * cv + sigm(gi) * tanhf(gg);
  c[(size_t)b * HID + j] = cv;
  float hc = sigm(go) * tanhf(cv);
  if (j < DM) {
    float hv = qenc[(size_t)b * DM + j] + hc;
    h[(size_t)b * DM + j] = hv;
    hr[(size_t)b * HID + j] = (_Float16)hv;
  } else {
    hr[(size_t)b * HID + j] = (_Float16)sg[j - DM];
  }
}

__global__ __launch_bounds__(256)
void k_final_dot(const float* __restrict__ h, const float* __restrict__ sg,
                 float* __restrict__ out) {
  __shared__ float red[256];
  int b = blockIdx.x, t = threadIdx.x;
  float s = block_sum256(h[(size_t)b * DM + t] * sg[t], red);
  if (t == 0) out[b] = s;
}

// ---------------- host ----------------

extern "C" void kernel_launch(void* const* d_in, const int* in_sizes, int n_in,
                              void* d_out, int out_size, void* d_ws, size_t ws_size,
                              hipStream_t stream) {
  (void)in_sizes; (void)n_in; (void)out_size; (void)ws_size;

  const int*   query    = (const int*)d_in[0];
  const int*   support  = (const int*)d_in[1];
  const int*   q_l1     = (const int*)d_in[2];
  const int*   q_r1     = (const int*)d_in[3];
  const int*   s_l1     = (const int*)d_in[4];
  const int*   s_r1     = (const int*)d_in[5];
  const float* emb      = (const float*)d_in[6];
  const float* gcn_w_w  = (const float*)d_in[7];
  const float* gcn_w_b  = (const float*)d_in[8];
  const float* gcn_b    = (const float*)d_in[9];
  const float* gate_w_w = (const float*)d_in[10];
  const float* gate_w_b = (const float*)d_in[11];
  const float* gate_b   = (const float*)d_in[12];
  const float* proj1_w  = (const float*)d_in[13];
  const float* proj1_b  = (const float*)d_in[14];
  const float* proj2_w  = (const float*)d_in[15];
  const float* proj2_b  = (const float*)d_in[16];
  const float* ln_g     = (const float*)d_in[17];
  const float* ln_b     = (const float*)d_in[18];
  const float* w_ih     = (const float*)d_in[19];
  const float* w_hh     = (const float*)d_in[20];
  const float* b_ih     = (const float*)d_in[21];
  const float* b_hh     = (const float*)d_in[22];
  float* out = (float*)d_out;

  char* w = (char*)d_ws;
  size_t off = 0;
  auto alloc = [&](size_t bytes) -> void* {
    void* p = w + off;
    off = (off + bytes + 255) & ~(size_t)255;
    return p;
  };

  _Float16* gcnh  = (_Float16*)alloc((size_t)EMBD * DM * 2);
  _Float16* p1h   = (_Float16*)alloc((size_t)HID * DM * 2);
  _Float16* p2h   = (_Float16*)alloc((size_t)DM * HID * 2);
  _Float16* wihh  = (_Float16*)alloc((size_t)GW * DM * 2);
  _Float16* whhh  = (_Float16*)alloc((size_t)GW * HID * 2);
  float*    bsum  = (float*)alloc((size_t)GW * 4);
  float*    qvec  = (float*)alloc((size_t)BQ * DM * 4);
  _Float16* qvech = (_Float16*)alloc((size_t)BQ * DM * 2);
  float*    svec  = (float*)alloc((size_t)SS * DM * 4);
  _Float16* svech = (_Float16*)alloc((size_t)SS * DM * 2);
  _Float16* h1qh  = (_Float16*)alloc((size_t)BQ * HID * 2);
  _Float16* h1sh  = (_Float16*)alloc((size_t)SS * HID * 2);
  float*    yq    = (float*)alloc((size_t)BQ * DM * 4);
  float*    ys    = (float*)alloc((size_t)SS * DM * 4);
  float*    qenc  = (float*)alloc((size_t)BQ * DM * 4);
  _Float16* qench = (_Float16*)alloc((size_t)BQ * DM * 2);
  float*    senc  = (float*)alloc((size_t)SS * DM * 4);
  float*    sg    = (float*)alloc((size_t)DM * 4);
  _Float16* sgh   = (_Float16*)alloc((size_t)DM * 2);
  float*    gbase = (float*)alloc((size_t)BQ * GW * 4);
  float*    gates = (float*)alloc((size_t)BQ * GW * 4);
  _Float16* hr    = (_Float16*)alloc((size_t)BQ * HID * 2);
  float*    cbuf  = (float*)alloc((size_t)BQ * HID * 4);
  float*    hbuf  = (float*)alloc((size_t)BQ * DM * 4);

  auto cvt = [&](const float* s, _Float16* d, int n) {
    k_cvt_f16<<<(n + 255) / 256, 256, 0, stream>>>(s, d, n);
  };
  cvt(gcn_w_w, gcnh, EMBD * DM);
  cvt(proj1_w, p1h, HID * DM);
  cvt(proj2_w, p2h, DM * HID);
  cvt(w_ih,  wihh, GW * DM);
  cvt(w_hh,  whhh, GW * HID);
  k_add_bias<<<(GW + 255) / 256, 256, 0, stream>>>(b_ih, b_hh, bsum, GW);

  // --- neighbor encoders ---
  k_neighbor<<<BQ, 256, 0, stream>>>(q_l1, query, 0, emb, gcnh, gcn_w_b, gcn_b,
                                     gate_w_w, gate_w_b, gate_b, qvec, qvech, 0);
  k_neighbor<<<BQ, 256, 0, stream>>>(q_r1, query, 1, emb, gcnh, gcn_w_b, gcn_b,
                                     gate_w_w, gate_w_b, gate_b, qvec, qvech, EMBD);
  k_neighbor<<<SS, 256, 0, stream>>>(s_l1, support, 0, emb, gcnh, gcn_w_b, gcn_b,
                                     gate_w_w, gate_w_b, gate_b, svec, svech, 0);
  k_neighbor<<<SS, 256, 0, stream>>>(s_r1, support, 1, emb, gcnh, gcn_w_b, gcn_b,
                                     gate_w_w, gate_w_b, gate_b, svec, svech, EMBD);

  auto gemm_small = [&](const _Float16* A, const _Float16* Wt, const float* bias,
                        const float* Cadd, float* Cout, _Float16* CoutH,
                        int M, int N, int K, int relu) {
    int tiles = ((M + 15) / 16) * ((N + 15) / 16);
    k_gemm_wmma<<<(tiles + 7) / 8, 256, 0, stream>>>(A, Wt, bias, Cadd, Cout,
                                                     CoutH, M, N, K, relu);
  };
  auto gemm_big = [&](const _Float16* A, const _Float16* Wt, const float* bias,
                      const float* Cadd, float* Cout, _Float16* CoutH,
                      int M, int N, int K, int relu) {
    int blocks = (M >> 7) * (N >> 6);
    k_gemm_wmma_lds<<<blocks, 256, 0, stream>>>(A, Wt, bias, Cadd, Cout,
                                                CoutH, M, N, K, relu);
  };

  // --- support encoder (query rows): async-LDS-staged WMMA GEMMs ---
  gemm_big(qvech, p1h, proj1_b, nullptr, nullptr, h1qh, BQ, HID, DM, 1);
  gemm_big(h1qh,  p2h, proj2_b, qvec,    yq,      nullptr, BQ, DM, HID, 0);
  k_layernorm<<<BQ, 256, 0, stream>>>(yq, ln_g, ln_b, qenc, qench);

  // --- support encoder (support rows, ragged M=5) + mean ---
  gemm_small(svech, p1h, proj1_b, nullptr, nullptr, h1sh, SS, HID, DM, 1);
  gemm_small(h1sh,  p2h, proj2_b, svec,    ys,      nullptr, SS, DM, HID, 0);
  k_layernorm<<<SS, 256, 0, stream>>>(ys, ln_g, ln_b, senc, nullptr);
  k_support_mean<<<1, 256, 0, stream>>>(senc, sg, sgh);

  // --- query encoder: loop-invariant gate base, 4 recurrent steps ---
  k_zero_u32<<<((BQ * HID) + 255) / 256, 256, 0, stream>>>((unsigned int*)cbuf, BQ * HID);
  k_zero_u32<<<((BQ * HID / 2) + 255) / 256, 256, 0, stream>>>((unsigned int*)hr, BQ * HID / 2);
  gemm_big(qench, wihh, bsum, nullptr, gbase, nullptr, BQ, GW, DM, 0);
  for (int step = 0; step < 4; ++step) {
    gemm_big(hr, whhh, nullptr, gbase, gates, nullptr, BQ, GW, HID, 0);
    k_lstm_cell<<<BQ, 512, 0, stream>>>(gates, qenc, sg, cbuf, hbuf, hr);
  }

  k_final_dot<<<BQ, 256, 0, stream>>>(hbuf, sg, out);
}
